// AttentionNet_32736240730545
// MI455X (gfx1250) — matrix operand order repaired and checked by reading, not compiled, not run on hardware
//
#include <hip/hip_runtime.h>

typedef __attribute__((ext_vector_type(16))) __bf16 bf16x16;
typedef __attribute__((ext_vector_type(8)))  float  f32x8;

// ---------------- helpers ----------------
__device__ __forceinline__ __bf16 f2bf(float x) {
  union { float f; unsigned u; } v; v.f = x;
  unsigned r = v.u + 0x7fffu + ((v.u >> 16) & 1u);
  union { unsigned short s; __bf16 b; } o; o.s = (unsigned short)(r >> 16);
  return o.b;
}
__device__ __forceinline__ float bf2f(__bf16 x) {
  union { unsigned short s; __bf16 b; } i; i.b = x;
  union { unsigned u; float f; } o; o.u = ((unsigned)i.s) << 16;
  return o.f;
}
__device__ __forceinline__ float sigmoidf_(float x) { return 1.f / (1.f + __expf(-x)); }

__device__ __forceinline__ f32x8 wmma_bf16(bf16x16 a, bf16x16 b, f32x8 c) {
  return __builtin_amdgcn_wmma_f32_16x16x32_bf16(false, a, false, b, (short)0, c, false, false);
}

// Fragment load: lane's 16 bf16 operand values are contiguous (32B) in swizzled tiles.
__device__ __forceinline__ bf16x16 ld_frag(const __bf16* p) {
  return *(const bf16x16*)p;
}

// Swizzled-tile element index helpers (512 elements per 16x32 A tile / 32x16 B tile).
// A (16x32, ISA layout): lane = m + 16*bit3(k), slot = 8*bit4(k) + (k&7)
// B (32x16, ISA layout): lane = n + 16*bit3(k), slot = 8*bit4(k) + (k&7)
__device__ __forceinline__ int a_swz(int m, int k) {
  return ((m & 15) + (((k >> 3) & 1) << 4)) * 16 + (((k >> 4) & 1) << 3) + (k & 7);
}
__device__ __forceinline__ int b_swz(int k, int n) {
  return ((n & 15) + (((k >> 3) & 1) << 4)) * 16 + (((k >> 4) & 1) << 3) + (k & 7);
}
// GRU hidden-state LDS index: tiles (b/16, i/32) of 512, swizzled as A.
template <int H>
__device__ __forceinline__ int h_idx(int b, int i) {
  return ((b >> 4) * (H >> 5) + (i >> 5)) * 512 + a_swz(b & 15, i & 31);
}

// ---------------- small utility kernels ----------------
// Convert f32 weight [K,N] -> bf16 in B-fragment swizzled tile layout.
// tile index = (k/32)*(N/16) + n/16, 512 elements per tile.
__global__ void cvt_swz_kernel(const float* __restrict__ in, __bf16* __restrict__ out,
                               int K, int N) {
  int i = blockIdx.x * 256 + threadIdx.x;
  if (i >= K * N) return;
  int k = i / N, n = i % N;
  size_t tile = (size_t)(k >> 5) * (N >> 4) + (n >> 4);
  out[tile * 512 + b_swz(k & 31, n)] = f2bf(in[i]);
}

__global__ void zero_kernel(float* p, int n) {
  int i = blockIdx.x * 256 + threadIdx.x;
  if (i < n) p[i] = 0.f;
}

__global__ void gather_kernel(const int* __restrict__ idx, const float* __restrict__ emb,
                              __bf16* __restrict__ out) {
  int row = blockIdx.x;
  int e = threadIdx.x; // 256 threads == E
  out[(size_t)row * 256 + e] = f2bf(emb[(size_t)idx[row] * 256 + e]);
}

__global__ void ecwm_kernel(const float* __restrict__ ec, const float* __restrict__ wm,
                            __bf16* __restrict__ out, int total) {
  int i = blockIdx.x * 256 + threadIdx.x;
  if (i < total) out[i] = f2bf(ec[i] * wm[i & 511]);
}

__global__ __launch_bounds__(256)
void rowdot_kernel(const float* __restrict__ X, const float* __restrict__ w,
                   float* __restrict__ out, int rows, int cols) {
  int wave = threadIdx.x >> 5, lane = threadIdx.x & 31;
  int row = blockIdx.x * 8 + wave;
  if (row >= rows) return;
  float acc = 0.f;
  for (int c = lane; c < cols; c += 32) acc += X[(size_t)row * cols + c] * w[c];
#pragma unroll
  for (int m = 16; m >= 1; m >>= 1) acc += __shfl_xor(acc, m, 32);
  if (lane == 0) out[row] = acc;
}

// ---------------- generic GEMM:  C[M,N] = A_bf16[M,K] @ Wsw[K,N] + bias ----------------
// block = 256 threads (8 waves), tile 128x64. A staged swizzled in LDS (double-buffered);
// B fragments loaded directly from pre-swizzled global weights (L2-resident).
__global__ __launch_bounds__(256)
void gemm_bias_kernel(const __bf16* __restrict__ A, const __bf16* __restrict__ Wsw,
                      const float* __restrict__ bias, float* __restrict__ C,
                      int K, int N) {
  __shared__ __align__(32) __bf16 sA[2][8 * 512];
  int tid = threadIdx.x, wave = tid >> 5, lane = tid & 31;
  size_t row0 = (size_t)blockIdx.y * 128;
  int col0 = blockIdx.x * 64;
  int ntiles = N >> 4;
  f32x8 acc[4] = {};

  auto stage = [&](int buf, int k0) {
#pragma unroll
    for (int ii = 0; ii < 2; ++ii) {
      int i = tid + ii * 256;
      int r = i >> 2, cb = (i & 3) << 3;
      uint4 v = *(const uint4*)(A + (row0 + r) * K + k0 + cb);
      int ln = (r & 15) + (((cb >> 3) & 1) << 4);
      *(uint4*)(&sA[buf][(r >> 4) * 512 + ln * 16 + ((cb >> 4) << 3)]) = v;
    }
  };

  stage(0, 0);
  __syncthreads();
  int cur = 0;
  for (int k0 = 0; k0 < K; k0 += 32) {
    if (k0 + 32 < K) stage(cur ^ 1, k0 + 32);           // overlap next-chunk fetch
    if (k0 + 64 < K)                                    // emits global_prefetch_b8
      __builtin_prefetch(A + (row0 + (tid >> 1)) * K + k0 + 64, 0, 1);
    bf16x16 af = ld_frag(&sA[cur][wave * 512 + lane * 16]);
    const __bf16* wb = Wsw + ((size_t)(k0 >> 5) * ntiles + (col0 >> 4)) * 512 + lane * 16;
    bf16x16 b0 = ld_frag(wb);
    bf16x16 b1 = ld_frag(wb + 512);
    bf16x16 b2 = ld_frag(wb + 1024);
    bf16x16 b3 = ld_frag(wb + 1536);
    acc[0] = wmma_bf16(af, b0, acc[0]);
    acc[1] = wmma_bf16(af, b1, acc[1]);
    acc[2] = wmma_bf16(af, b2, acc[2]);
    acc[3] = wmma_bf16(af, b3, acc[3]);
    __syncthreads();
    cur ^= 1;
  }
  int n = lane & 15;
  int rbase = wave * 16 + ((lane < 16) ? 0 : 8);
#pragma unroll
  for (int tn = 0; tn < 4; ++tn) {
    int col = col0 + tn * 16 + n;
    float bv = bias ? bias[col] : 0.f;
#pragma unroll
    for (int r = 0; r < 8; ++r)
      C[(row0 + rbase + r) * N + col] = acc[tn][r] + bv;
  }
}

// ---------------- GRU recurrence (both directions in one grid) ----------------
// One 16-wave block per direction; h kept in LDS (bf16, swizzled A layout);
// hg = h@Wh via WMMA with B fragments from pre-swizzled global Wh.
template <int H>
__global__ __launch_bounds__(512)
void gru_rec_kernel(const float* __restrict__ xg_f, const float* __restrict__ xg_b,
                    const __bf16* __restrict__ Wh_fw, const __bf16* __restrict__ Wh_bw,
                    const float* __restrict__ bh_fw, const float* __restrict__ bh_bw,
                    float* out_f32, __bf16* out_bf, int S, int ostride) {
  constexpr int PW = H / 64; // tile-tasks per wave (16 waves cover 4 x H/16 tiles)
  int dir = blockIdx.x;
  const float* xg = dir ? xg_b : xg_f;
  const __bf16* Wh = dir ? Wh_bw : Wh_fw;
  const float* bh = dir ? bh_bw : bh_fw;
  int ooff = dir * H;
  __shared__ __align__(32) __bf16 hbuf[64 * H];
  int tid = threadIdx.x, wave = tid >> 5, lane = tid & 31;
  for (int i = tid; i < 64 * H; i += 512) hbuf[i] = f2bf(0.f);
  __syncthreads();
  int n = lane & 15;
  int hi8 = (lane < 16) ? 0 : 8;
  for (int t = 0; t < S; ++t) {
    int p = dir ? (S - 1 - t) : t; // reversed scan for backward dir
    float hnew[PW][8];
#pragma unroll
    for (int ti = 0; ti < PW; ++ti) {
      int task = wave * PW + ti;
      int tm = task / (H / 16);
      int tn = task % (H / 16);
      f32x8 cr = {}, cz = {}, cn = {};
      for (int k0 = 0; k0 < H; k0 += 32) {
        bf16x16 af = ld_frag(&hbuf[((tm * (H >> 5)) + (k0 >> 5)) * 512 + lane * 16]);
        const __bf16* wrow = Wh + (size_t)(k0 >> 5) * ((3 * H) >> 4) * 512 + (size_t)lane * 16;
        bf16x16 br = ld_frag(wrow + (size_t)(0 * (H >> 4) + tn) * 512);
        bf16x16 bz = ld_frag(wrow + (size_t)(1 * (H >> 4) + tn) * 512);
        bf16x16 bn = ld_frag(wrow + (size_t)(2 * (H >> 4) + tn) * 512);
        cr = wmma_bf16(af, br, cr);
        cz = wmma_bf16(af, bz, cz);
        cn = wmma_bf16(af, bn, cn);
      }
      int col = tn * 16 + n;
      float bhr = bh[col], bhz = bh[H + col], bhn = bh[2 * H + col];
#pragma unroll
      for (int r = 0; r < 8; ++r) {
        int b = tm * 16 + hi8 + r;
        size_t xrow = ((size_t)b * S + p) * (size_t)(3 * H);
        float xr = xg[xrow + col];
        float xz = xg[xrow + H + col];
        float xn = xg[xrow + 2 * H + col];
        float hv = bf2f(hbuf[h_idx<H>(b, col)]);
        float rg = sigmoidf_(xr + cr[r] + bhr);
        float zg = sigmoidf_(xz + cz[r] + bhz);
        float ng = tanhf(xn + rg * (cn[r] + bhn));
        hnew[ti][r] = (1.f - zg) * ng + zg * hv;
      }
    }
    __syncthreads(); // all reads of h done
#pragma unroll
    for (int ti = 0; ti < PW; ++ti) {
      int task = wave * PW + ti;
      int tm = task / (H / 16);
      int tn = task % (H / 16);
      int col = tn * 16 + n;
#pragma unroll
      for (int r = 0; r < 8; ++r) {
        int b = tm * 16 + hi8 + r;
        float v = hnew[ti][r];
        hbuf[h_idx<H>(b, col)] = f2bf(v);
        size_t o = ((size_t)b * S + p) * (size_t)ostride + ooff + col;
        if (out_f32) out_f32[o] = v;
        if (out_bf) out_bf[o] = f2bf(v);
      }
    }
    __syncthreads();
  }
}

// ---------------- attention scores + row softmax over J (=64, fits one block) ----------------
__global__ __launch_bounds__(256)
void attn_score_kernel(const __bf16* __restrict__ ecwm, const __bf16* __restrict__ eqb,
                       const float* __restrict__ s_c, const float* __restrict__ s_q,
                       __bf16* __restrict__ S_bf, float* __restrict__ msoft) {
  __shared__ __align__(32) __bf16 sA[2][8 * 512];
  __shared__ __align__(32) __bf16 sBt[2][4 * 512]; // eq^T as B tiles (k=d, n=j)
  __shared__ float sS[128 * 64];
  int b = blockIdx.y;
  int row0 = blockIdx.x * 128;
  int tid = threadIdx.x, wave = tid >> 5, lane = tid & 31;
  f32x8 acc[4] = {};

  auto stage = [&](int buf, int k0) {
#pragma unroll
    for (int ii = 0; ii < 2; ++ii) { // A: 128 rows x 4 chunks
      int i = tid + ii * 256;
      int r = i >> 2, cb = (i & 3) << 3;
      uint4 v = *(const uint4*)(ecwm + ((size_t)b * 512 + row0 + r) * 512 + k0 + cb);
      int ln = (r & 15) + (((cb >> 3) & 1) << 4);
      *(uint4*)(&sA[buf][(r >> 4) * 512 + ln * 16 + ((cb >> 4) << 3)]) = v;
    }
    { // B^T: 64 j-rows x 4 chunks (B swizzle), exactly 256 uint4 chunks
      int r = tid >> 2, cb = (tid & 3) << 3;
      uint4 v = *(const uint4*)(eqb + ((size_t)b * 64 + r) * 512 + k0 + cb);
      int ln = (r & 15) + (((cb >> 3) & 1) << 4);
      *(uint4*)(&sBt[buf][(r >> 4) * 512 + ln * 16 + ((cb >> 4) << 3)]) = v;
    }
  };

  stage(0, 0);
  __syncthreads();
  int cur = 0;
  for (int k0 = 0; k0 < 512; k0 += 32) {
    if (k0 + 32 < 512) stage(cur ^ 1, k0 + 32);
    bf16x16 af = ld_frag(&sA[cur][wave * 512 + lane * 16]);
    bf16x16 b0 = ld_frag(&sBt[cur][0 * 512 + lane * 16]);
    bf16x16 b1 = ld_frag(&sBt[cur][1 * 512 + lane * 16]);
    bf16x16 b2 = ld_frag(&sBt[cur][2 * 512 + lane * 16]);
    bf16x16 b3 = ld_frag(&sBt[cur][3 * 512 + lane * 16]);
    acc[0] = wmma_bf16(af, b0, acc[0]);
    acc[1] = wmma_bf16(af, b1, acc[1]);
    acc[2] = wmma_bf16(af, b2, acc[2]);
    acc[3] = wmma_bf16(af, b3, acc[3]);
    __syncthreads();
    cur ^= 1;
  }
  int n = lane & 15;
  int rbase = wave * 16 + ((lane < 16) ? 0 : 8);
#pragma unroll
  for (int tn = 0; tn < 4; ++tn) {
    int j = tn * 16 + n;
    float sq = s_q[b * 64 + j];
#pragma unroll
    for (int r = 0; r < 8; ++r) {
      int rr = rbase + r;
      sS[rr * 64 + j] = acc[tn][r] + s_c[(size_t)b * 512 + row0 + rr] + sq;
    }
  }
  __syncthreads();
  if (tid < 128) {
    float m = -1e30f;
    for (int j = 0; j < 64; ++j) m = fmaxf(m, sS[tid * 64 + j]);
    float s = 0.f;
    for (int j = 0; j < 64; ++j) s += __expf(sS[tid * 64 + j] - m);
    float inv = 1.f / s;
    size_t rowg = (size_t)b * 512 + row0 + tid;
    for (int j = 0; j < 64; ++j)
      S_bf[rowg * 64 + j] = f2bf(__expf(sS[tid * 64 + j] - m) * inv);
    msoft[rowg] = inv; // == max_j softmax(row): exp(max-max)/sum
  }
}

// ---------------- c2q_pre[b,t,d] = sum_j S[b,t,j] * eq[b,j,d]  (batched WMMA GEMM) ----------------
__global__ __launch_bounds__(256)
void c2q_gemm_kernel(const __bf16* __restrict__ S_bf, const __bf16* __restrict__ eqb,
                     float* __restrict__ c2q_pre) {
  __shared__ __align__(32) __bf16 sA[8 * 512];
  __shared__ __align__(32) __bf16 sB[4 * 512];
  int b = blockIdx.z;
  int row0 = blockIdx.y * 128;
  int col0 = blockIdx.x * 64;
  int tid = threadIdx.x, wave = tid >> 5, lane = tid & 31;
  f32x8 acc[4] = {};
  for (int k0 = 0; k0 < 64; k0 += 32) {
#pragma unroll
    for (int ii = 0; ii < 2; ++ii) { // A: S rows, K=64
      int i = tid + ii * 256;
      int r = i >> 2, cb = (i & 3) << 3;
      uint4 v = *(const uint4*)(S_bf + ((size_t)b * 512 + row0 + r) * 64 + k0 + cb);
      int ln = (r & 15) + (((cb >> 3) & 1) << 4);
      *(uint4*)(sA + (r >> 4) * 512 + ln * 16 + ((cb >> 4) << 3)) = v;
    }
#pragma unroll
    for (int ii = 0; ii < 8; ++ii) { // B: eq rows k0..k0+31 x 64 cols (scatter swizzle)
      int i = tid + ii * 256;
      int kk = i >> 6, c = i & 63;
      __bf16 v = eqb[((size_t)b * 64 + k0 + kk) * 512 + col0 + c];
      sB[(c >> 4) * 512 + b_swz(kk, c & 15)] = v;
    }
    __syncthreads();
    bf16x16 af = ld_frag(sA + wave * 512 + lane * 16);
    bf16x16 b0 = ld_frag(sB + 0 * 512 + lane * 16);
    bf16x16 b1 = ld_frag(sB + 1 * 512 + lane * 16);
    bf16x16 b2 = ld_frag(sB + 2 * 512 + lane * 16);
    bf16x16 b3 = ld_frag(sB + 3 * 512 + lane * 16);
    acc[0] = wmma_bf16(af, b0, acc[0]);
    acc[1] = wmma_bf16(af, b1, acc[1]);
    acc[2] = wmma_bf16(af, b2, acc[2]);
    acc[3] = wmma_bf16(af, b3, acc[3]);
    __syncthreads();
  }
  int n = lane & 15;
  int rbase = wave * 16 + ((lane < 16) ? 0 : 8);
#pragma unroll
  for (int tn = 0; tn < 4; ++tn) {
    int col = col0 + tn * 16 + n;
#pragma unroll
    for (int r = 0; r < 8; ++r)
      c2q_pre[((size_t)b * 512 + row0 + rbase + r) * 512 + col] = acc[tn][r];
  }
}

// ---------------- row softmax / log-softmax ----------------
__global__ __launch_bounds__(256)
void softmax_rows_kernel(const float* __restrict__ in, float* __restrict__ out, int cols) {
  __shared__ float red[256];
  size_t base = (size_t)blockIdx.x * cols;
  int tid = threadIdx.x;
  float m = -1e30f;
  for (int c = tid; c < cols; c += 256) m = fmaxf(m, in[base + c]);
  red[tid] = m; __syncthreads();
  for (int s = 128; s > 0; s >>= 1) { if (tid < s) red[tid] = fmaxf(red[tid], red[tid + s]); __syncthreads(); }
  m = red[0]; __syncthreads();
  float sum = 0.f;
  for (int c = tid; c < cols; c += 256) sum += __expf(in[base + c] - m);
  red[tid] = sum; __syncthreads();
  for (int s = 128; s > 0; s >>= 1) { if (tid < s) red[tid] += red[tid + s]; __syncthreads(); }
  float inv = 1.f / red[0];
  for (int c = tid; c < cols; c += 256) out[base + c] = __expf(in[base + c] - m) * inv;
}

__global__ __launch_bounds__(256)
void logsoftmax_rows_kernel(const float* __restrict__ in, float* __restrict__ out, int cols) {
  __shared__ float red[256];
  size_t base = (size_t)blockIdx.x * cols;
  int tid = threadIdx.x;
  float m = -1e30f;
  for (int c = tid; c < cols; c += 256) m = fmaxf(m, in[base + c]);
  red[tid] = m; __syncthreads();
  for (int s = 128; s > 0; s >>= 1) { if (tid < s) red[tid] = fmaxf(red[tid], red[tid + s]); __syncthreads(); }
  m = red[0]; __syncthreads();
  float sum = 0.f;
  for (int c = tid; c < cols; c += 256) sum += __expf(in[base + c] - m);
  red[tid] = sum; __syncthreads();
  for (int s = 128; s > 0; s >>= 1) { if (tid < s) red[tid] += red[tid + s]; __syncthreads(); }
  float lse = m + logf(red[0]);
  for (int c = tid; c < cols; c += 256) out[base + c] = in[base + c] - lse;
}

// ---------------- q2c: softmax over t of msoft, then weighted sum of ec ----------------
__global__ __launch_bounds__(512)
void q2c_kernel(const float* __restrict__ msoft, const float* __restrict__ ec,
                float* __restrict__ q2c) {
  __shared__ float w[512];
  __shared__ float red[512];
  int b = blockIdx.x, tid = threadIdx.x;
  float x = msoft[(size_t)b * 512 + tid];
  red[tid] = x; __syncthreads();
  for (int s = 256; s > 0; s >>= 1) { if (tid < s) red[tid] = fmaxf(red[tid], red[tid + s]); __syncthreads(); }
  float m = red[0]; __syncthreads();
  float e = __expf(x - m);
  red[tid] = e; __syncthreads();
  for (int s = 256; s > 0; s >>= 1) { if (tid < s) red[tid] += red[tid + s]; __syncthreads(); }
  w[tid] = e / red[0];
  __syncthreads();
  float accum = 0.f;
  for (int t = 0; t < 512; ++t)
    accum += w[t] * ec[((size_t)b * 512 + t) * 512 + tid];
  q2c[(size_t)b * 512 + tid] = accum;
}

// ---------------- G = [ec, c2q, ec*c2q, ec*q2c] as bf16 ----------------
__global__ void build_G_kernel(const float* __restrict__ ec, const float* __restrict__ c2q,
                               const float* __restrict__ q2c, __bf16* __restrict__ G) {
  size_t i = (size_t)blockIdx.x * 256 + threadIdx.x;
  if (i >= (size_t)32768 * 512) return;
  size_t r = i >> 9;
  int d = (int)(i & 511);
  int b = (int)(r >> 9);
  float e = ec[i];
  float c = c2q[i];
  size_t g = r * 2048 + d;
  G[g] = f2bf(e);
  G[g + 512] = f2bf(c);
  G[g + 1024] = f2bf(e * c);
  G[g + 1536] = f2bf(e * q2c[(size_t)b * 512 + d]);
}

// ---------------- pointer net: u[row] = sum_n tanh((M@W1)[row,n]) * vt[n] ----------------
__global__ __launch_bounds__(256)
void ptr_gemm_kernel(const __bf16* __restrict__ Mb, const __bf16* __restrict__ W1sw,
                     const float* __restrict__ vt, float* __restrict__ u) {
  __shared__ __align__(32) __bf16 sA[2][8 * 512];
  const int K = 1024, N = 1024;
  int tid = threadIdx.x, wave = tid >> 5, lane = tid & 31;
  size_t row0 = (size_t)blockIdx.y * 128;
  int col0 = blockIdx.x * 64;
  f32x8 acc[4] = {};

  auto stage = [&](int buf, int k0) {
#pragma unroll
    for (int ii = 0; ii < 2; ++ii) {
      int i = tid + ii * 256;
      int r = i >> 2, cb = (i & 3) << 3;
      uint4 v = *(const uint4*)(Mb + (row0 + r) * K + k0 + cb);
      int ln = (r & 15) + (((cb >> 3) & 1) << 4);
      *(uint4*)(&sA[buf][(r >> 4) * 512 + ln * 16 + ((cb >> 4) << 3)]) = v;
    }
  };

  stage(0, 0);
  __syncthreads();
  int cur = 0;
  for (int k0 = 0; k0 < K; k0 += 32) {
    if (k0 + 32 < K) stage(cur ^ 1, k0 + 32);
    if (k0 + 64 < K)
      __builtin_prefetch(Mb + (row0 + (tid >> 1)) * K + k0 + 64, 0, 1);
    bf16x16 af = ld_frag(&sA[cur][wave * 512 + lane * 16]);
    const __bf16* wb = W1sw + ((size_t)(k0 >> 5) * (N >> 4) + (col0 >> 4)) * 512 + lane * 16;
    bf16x16 b0 = ld_frag(wb);
    bf16x16 b1 = ld_frag(wb + 512);
    bf16x16 b2 = ld_frag(wb + 1024);
    bf16x16 b3 = ld_frag(wb + 1536);
    acc[0] = wmma_bf16(af, b0, acc[0]);
    acc[1] = wmma_bf16(af, b1, acc[1]);
    acc[2] = wmma_bf16(af, b2, acc[2]);
    acc[3] = wmma_bf16(af, b3, acc[3]);
    __syncthreads();
    cur ^= 1;
  }
  int n = lane & 15;
  int rbase = wave * 16 + ((lane < 16) ? 0 : 8);
  float pr[8] = {};
#pragma unroll
  for (int tn = 0; tn < 4; ++tn) {
    float v = vt[col0 + tn * 16 + n];
#pragma unroll
    for (int r = 0; r < 8; ++r) pr[r] += tanhf(acc[tn][r]) * v;
  }
#pragma unroll
  for (int r = 0; r < 8; ++r) atomicAdd(&u[row0 + rbase + r], pr[r]);
}

// ---------------- host side ----------------
extern "C" void kernel_launch(void* const* d_in, const int* in_sizes, int n_in,
                              void* d_out, int out_size, void* d_ws, size_t ws_size,
                              hipStream_t stream) {
  (void)in_sizes; (void)n_in; (void)out_size; (void)ws_size;
  const int*   context = (const int*)d_in[0];
  const int*   query   = (const int*)d_in[1];
  const float* emb     = (const float*)d_in[2];
  const float* ctxWi_f = (const float*)d_in[3];
  const float* ctxWh_f = (const float*)d_in[4];
  const float* ctxbi_f = (const float*)d_in[5];
  const float* ctxbh_f = (const float*)d_in[6];
  const float* ctxWi_b = (const float*)d_in[7];
  const float* ctxWh_b = (const float*)d_in[8];
  const float* ctxbi_b = (const float*)d_in[9];
  const float* ctxbh_b = (const float*)d_in[10];
  const float* W_att   = (const float*)d_in[11];
  const float* modWi_f = (const float*)d_in[12];
  const float* modWh_f = (const float*)d_in[13];
  const float* modbi_f = (const float*)d_in[14];
  const float* modbh_f = (const float*)d_in[15];
  const float* modWi_b = (const float*)d_in[16];
  const float* modWh_b = (const float*)d_in[17];
  const float* modbi_b = (const float*)d_in[18];
  const float* modbh_b = (const float*)d_in[19];
  const float* ptrW1   = (const float*)d_in[20];
  const float* ptrvt   = (const float*)d_in[21];

  float* out_logp = (float*)d_out;              // [B,1,T] = 64*512
  float* out_c2q  = (float*)d_out + 64 * 512;   // [B,T,D] = 32768*512

  char* ws = (char*)d_ws;
  size_t off = 0;
  auto alloc = [&](size_t bytes) -> char* {
    char* p = ws + off;
    off += (bytes + 255) & ~(size_t)255;
    return p;
  };
  __bf16* xctx_bf  = (__bf16*)alloc((size_t)32768 * 256 * 2);
  __bf16* xqry_bf  = (__bf16*)alloc((size_t)4096 * 256 * 2);
  float*  xgctx_f  = (float*)alloc((size_t)32768 * 768 * 4);
  float*  xgctx_b  = (float*)alloc((size_t)32768 * 768 * 4);
  float*  xgqry_f  = (float*)alloc((size_t)4096 * 768 * 4);
  float*  xgqry_b  = (float*)alloc((size_t)4096 * 768 * 4);
  float*  ec       = (float*)alloc((size_t)32768 * 512 * 4);
  float*  eq       = (float*)alloc((size_t)4096 * 512 * 4);
  __bf16* eq_bf    = (__bf16*)alloc((size_t)4096 * 512 * 2);
  __bf16* ecwm_bf  = (__bf16*)alloc((size_t)32768 * 512 * 2);
  float*  s_c      = (float*)alloc((size_t)32768 * 4);
  float*  s_q      = (float*)alloc((size_t)4096 * 4);
  __bf16* S_bf     = (__bf16*)alloc((size_t)32768 * 64 * 2);
  float*  msoft    = (float*)alloc((size_t)32768 * 4);
  float*  c2q_pre  = (float*)alloc((size_t)32768 * 512 * 4);
  float*  q2c      = (float*)alloc((size_t)64 * 512 * 4);
  __bf16* G_bf     = (__bf16*)alloc((size_t)32768 * 2048 * 2);
  float*  xgmod_f  = (float*)alloc((size_t)32768 * 1536 * 4);
  float*  xgmod_b  = (float*)alloc((size_t)32768 * 1536 * 4);
  __bf16* M_bf     = (__bf16*)alloc((size_t)32768 * 1024 * 2);
  float*  u        = (float*)alloc((size_t)32768 * 4);
  __bf16* bWi_cf   = (__bf16*)alloc((size_t)256 * 768 * 2);
  __bf16* bWi_cb   = (__bf16*)alloc((size_t)256 * 768 * 2);
  __bf16* bWh_cf   = (__bf16*)alloc((size_t)256 * 768 * 2);
  __bf16* bWh_cb   = (__bf16*)alloc((size_t)256 * 768 * 2);
  __bf16* bWi_mf   = (__bf16*)alloc((size_t)2048 * 1536 * 2);
  __bf16* bWi_mb   = (__bf16*)alloc((size_t)2048 * 1536 * 2);
  __bf16* bWh_mf   = (__bf16*)alloc((size_t)512 * 1536 * 2);
  __bf16* bWh_mb   = (__bf16*)alloc((size_t)512 * 1536 * 2);
  __bf16* bW1      = (__bf16*)alloc((size_t)1024 * 1024 * 2);

  auto cvtsw = [&](const float* s, __bf16* dpt, int K, int N) {
    cvt_swz_kernel<<<(K * N + 255) / 256, 256, 0, stream>>>(s, dpt, K, N);
  };
  cvtsw(ctxWi_f, bWi_cf, 256, 768);
  cvtsw(ctxWi_b, bWi_cb, 256, 768);
  cvtsw(ctxWh_f, bWh_cf, 256, 768);
  cvtsw(ctxWh_b, bWh_cb, 256, 768);
  cvtsw(modWi_f, bWi_mf, 2048, 1536);
  cvtsw(modWi_b, bWi_mb, 2048, 1536);
  cvtsw(modWh_f, bWh_mf, 512, 1536);
  cvtsw(modWh_b, bWh_mb, 512, 1536);
  cvtsw(ptrW1,   bW1,    1024, 1024);

  gather_kernel<<<32768, 256, 0, stream>>>(context, emb, xctx_bf);
  gather_kernel<<<4096, 256, 0, stream>>>(query, emb, xqry_bf);

  // input gates xg = x @ Wi + bi (big parallel WMMA GEMMs)
  gemm_bias_kernel<<<dim3(12, 256), 256, 0, stream>>>(xctx_bf, bWi_cf, ctxbi_f, xgctx_f, 256, 768);
  gemm_bias_kernel<<<dim3(12, 256), 256, 0, stream>>>(xctx_bf, bWi_cb, ctxbi_b, xgctx_b, 256, 768);
  gemm_bias_kernel<<<dim3(12, 32), 256, 0, stream>>>(xqry_bf, bWi_cf, ctxbi_f, xgqry_f, 256, 768);
  gemm_bias_kernel<<<dim3(12, 32), 256, 0, stream>>>(xqry_bf, bWi_cb, ctxbi_b, xgqry_b, 256, 768);

  // contextual BiGRU scans (one block per direction)
  gru_rec_kernel<256><<<2, 512, 0, stream>>>(xgctx_f, xgctx_b, bWh_cf, bWh_cb, ctxbh_f, ctxbh_b,
                                             ec, (__bf16*)nullptr, 512, 512);
  gru_rec_kernel<256><<<2, 512, 0, stream>>>(xgqry_f, xgqry_b, bWh_cf, bWh_cb, ctxbh_f, ctxbh_b,
                                             eq, eq_bf, 64, 512);

  // attention
  ecwm_kernel<<<(32768 * 512 + 255) / 256, 256, 0, stream>>>(ec, W_att + 1024, ecwm_bf, 32768 * 512);
  rowdot_kernel<<<32768 / 8, 256, 0, stream>>>(ec, W_att, s_c, 32768, 512);
  rowdot_kernel<<<4096 / 8, 256, 0, stream>>>(eq, W_att + 512, s_q, 4096, 512);
  attn_score_kernel<<<dim3(4, 64), 256, 0, stream>>>(ecwm_bf, eq_bf, s_c, s_q, S_bf, msoft);
  c2q_gemm_kernel<<<dim3(8, 4, 64), 256, 0, stream>>>(S_bf, eq_bf, c2q_pre);
  softmax_rows_kernel<<<32768, 256, 0, stream>>>(c2q_pre, out_c2q, 512);
  q2c_kernel<<<64, 512, 0, stream>>>(msoft, ec, q2c);
  build_G_kernel<<<(32768 * 512 + 255) / 256, 256, 0, stream>>>(ec, out_c2q, q2c, G_bf);

  // modeling BiGRU
  gemm_bias_kernel<<<dim3(24, 256), 256, 0, stream>>>(G_bf, bWi_mf, modbi_f, xgmod_f, 2048, 1536);
  gemm_bias_kernel<<<dim3(24, 256), 256, 0, stream>>>(G_bf, bWi_mb, modbi_b, xgmod_b, 2048, 1536);
  gru_rec_kernel<512><<<2, 512, 0, stream>>>(xgmod_f, xgmod_b, bWh_mf, bWh_mb, modbh_f, modbh_b,
                                             (float*)nullptr, M_bf, 512, 1024);

  // pointer net + log-softmax
  zero_kernel<<<32768 / 256, 256, 0, stream>>>(u, 32768);
  ptr_gemm_kernel<<<dim3(16, 256), 256, 0, stream>>>(M_bf, bW1, ptrvt, u);
  logsoftmax_rows_kernel<<<64, 256, 0, stream>>>(u, out_logp, 512);
}
